// AdvancedConnectomeGNN_26766236188923
// MI455X (gfx1250) — matrix operand order repaired and checked by reading, not compiled, not run on hardware
//
#include <hip/hip_runtime.h>
#include <hip/hip_bf16.h>

// ---------------------------------------------------------------------------
// AdvancedConnectomeGNN forward on MI455X (gfx1250, wave32, WMMA).
//
// All matmuls in f16 via v_wmma_f32_16x16x32_f16 with f32 accumulation.
// Edge MLP (~34 GFLOP/layer, dominant) fuses the gather
// (A row = concat(h[dst],h[src])) into LDS staging and fuses segment_sum into
// the epilogue with global_atomic_add_f32. Attention is flash-style streaming.
// Weights are pre-transposed during the f32->f16 conversion so both A and B
// LDS staging are pure 16B load/ds_store_b128 paths (no scalar transposes in
// the hot loops); V is pre-transposed once per MHA for the same reason.
// ---------------------------------------------------------------------------

typedef __attribute__((ext_vector_type(16))) _Float16 v16h;
typedef __attribute__((ext_vector_type(8)))  _Float16 v8h;
typedef __attribute__((ext_vector_type(8)))  float    v8f;

static constexpr int kN = 2048, kE = 131072, kF = 128, kD = 256, kL = 4, kH = 8, kB = 4;

__device__ __forceinline__ v8f wmma16(v16h a, v16h b, v8f c) {
  // D = A(16x32 f16) x B(32x16 f16) + C(16x16 f32)
  return __builtin_amdgcn_wmma_f32_16x16x32_f16(false, a, false, b, (short)0, c, false, false);
}

// A-fragment (16x32 f16): lane m=lane&15, kbase=(lane<16)?0:8.
// halves e<8 at k=kbase+e, e>=8 at k=kbase+16+(e-8)  -> two 16B loads.
__device__ __forceinline__ v16h load_afrag(const _Float16* row, int kb) {
  v8h lo = *(const v8h*)(row + kb);
  v8h hi = *(const v8h*)(row + kb + 16);
  v16h r;
#pragma unroll
  for (int i = 0; i < 8; ++i) { r[i] = lo[i]; r[i + 8] = hi[i]; }
  return r;
}

// B-fragment (32x16 f16): lane n=lane&15, kbase=(lane<16)?0:8,
// e<8 at k=kbase+e, e>=8 at k=kbase+16+(e-8) -> two 16B loads from a K-contiguous column.
__device__ __forceinline__ v16h load_bfrag(const _Float16* col, int kb) {
  v8h lo = *(const v8h*)(col + kb);
  v8h hi = *(const v8h*)(col + kb + 16);
  v16h r;
#pragma unroll
  for (int i = 0; i < 8; ++i) { r[i] = lo[i]; r[i + 8] = hi[i]; }
  return r;
}

// ---------------------------------------------------------------------------
// Generic WMMA GEMM:  C[M,Nc] = act( Arow @ B + bias ) * rowscale + residual
// A row m = concat( X[i0? i0[m]:m][0:Ksplit], Y[i1? i1[m]:m][0:K-Ksplit] )
// Bt is the PRE-TRANSPOSED weight: Bt[n][k]  (layout [Nc][K], f16).
// SCATTER: atomicAdd rows into C[sidx[m]] (segment_sum fusion).
// Tiles: 64x64x32, 128 threads (4 waves), each wave a 32x32 subtile
// = 4 wmma per K-step.
// ---------------------------------------------------------------------------
#define LDS_STRIDE 48   // 32 + 16 halves pad: 96B rows keep 16B alignment

template <int ACT, bool SCATTER>
__global__ void __launch_bounds__(128)
gemm_wmma_k(const _Float16* __restrict__ X, const _Float16* __restrict__ Y,
            int Ksplit, const int* __restrict__ i0, const int* __restrict__ i1,
            const _Float16* __restrict__ Bt, const float* __restrict__ bias,
            const float* __restrict__ rowscale, const float* __restrict__ residual,
            float* __restrict__ C, _Float16* __restrict__ C16,
            const int* __restrict__ sidx, int M, int K, int Nc) {
  __shared__ __align__(16) _Float16 As[64 * LDS_STRIDE];
  __shared__ __align__(16) _Float16 Bs[64 * LDS_STRIDE];

  const int tid  = threadIdx.x;
  const int lane = tid & 31, wave = tid >> 5;
  const int nb = blockIdx.x * 64, mb = blockIdx.y * 64;
  const int wr = (wave >> 1) * 32, wc = (wave & 1) * 32;

  // staging roles: 2 threads/row, 16 halves (two 16B chunks) each
  const int sr = tid >> 1, spart = tid & 1;
  const int am = mb + sr;
  const int ia0 = i0 ? i0[am] : am;
  const int ia1 = i1 ? i1[am] : am;
  const int KY  = K - Ksplit;
  const _Float16* brow = Bt + (size_t)(nb + sr) * K;   // transposed weight row

  v8f acc[4] = {};

  for (int kk = 0; kk < K; kk += 32) {
    const int ks = kk + spart * 16;
    { // stage A tile (optional 2-source gather), vectorized
      const _Float16* src = (ks < Ksplit)
          ? (X + (size_t)ia0 * Ksplit + ks)
          : (Y + (size_t)ia1 * KY + (ks - Ksplit));
      *(v8h*)(&As[sr * LDS_STRIDE + spart * 16])     = *(const v8h*)(src);
      *(v8h*)(&As[sr * LDS_STRIDE + spart * 16 + 8]) = *(const v8h*)(src + 8);
    }
    { // stage Bt tile, vectorized (already [n][k])
      const _Float16* src = brow + ks;
      *(v8h*)(&Bs[sr * LDS_STRIDE + spart * 16])     = *(const v8h*)(src);
      *(v8h*)(&Bs[sr * LDS_STRIDE + spart * 16 + 8]) = *(const v8h*)(src + 8);
      if (kk + 32 < K)  // -> global_prefetch_b8
        __builtin_prefetch(brow + ks + 32, 0, 1);
    }
    __syncthreads();

    const int akb = (lane < 16) ? 0 : 8;
    const int bkb = (lane < 16) ? 0 : 8;
    v16h a0 = load_afrag(&As[(wr + (lane & 15)) * LDS_STRIDE], akb);
    v16h a1 = load_afrag(&As[(wr + 16 + (lane & 15)) * LDS_STRIDE], akb);
    v16h b0 = load_bfrag(&Bs[(wc + (lane & 15)) * LDS_STRIDE], bkb);
    v16h b1 = load_bfrag(&Bs[(wc + 16 + (lane & 15)) * LDS_STRIDE], bkb);
    acc[0] = wmma16(a0, b0, acc[0]);
    acc[1] = wmma16(a0, b1, acc[1]);
    acc[2] = wmma16(a1, b0, acc[2]);
    acc[3] = wmma16(a1, b1, acc[3]);
    __syncthreads();
  }

  // epilogue: C layout -> lane col = lane&15, rows v (+8 for lanes>=16)
  const int rb = (lane >= 16) ? 8 : 0;
  const int cl = lane & 15;
#pragma unroll
  for (int ci = 0; ci < 4; ++ci) {
    const int rsub = wr + (ci >> 1) * 16;
    const int csub = wc + (ci & 1) * 16;
#pragma unroll
    for (int v = 0; v < 8; ++v) {
      const int gm = mb + rsub + rb + v;
      const int gn = nb + csub + cl;
      float val = acc[ci][v];
      if (bias) val += bias[gn];
      if (ACT == 1) val = val > 0.f ? val : 0.f;
      if (rowscale) val *= rowscale[gm];
      if (residual) val += residual[(size_t)gm * Nc + gn];
      if (SCATTER) {
        atomicAdd(&C[(size_t)sidx[gm] * Nc + gn], val);
      } else {
        C[(size_t)gm * Nc + gn] = val;
        if (C16) C16[(size_t)gm * Nc + gn] = (_Float16)val;
      }
    }
  }
}

// ---------------------------------------------------------------------------
// Flash attention, DH = head dim (32 for H=8, 64 for H=4). qkv layout [N,768]
// with q|k|v at cols 0/256/512. vT is the pre-transposed V section:
// vT[c][key] with c = head*DH + n (layout [256][N], f16) so V staging is
// vectorized. Block = 4 waves, each wave owns 16 query rows; 32-key chunks
// staged in LDS and shared by all 4 waves. Online softmax in the WMMA
// C-register layout with __shfl_xor row reductions over the 16-lane halves.
// ---------------------------------------------------------------------------
template <int DH>
__global__ void __launch_bounds__(128)
flash_attn_k(const _Float16* __restrict__ qkv, const _Float16* __restrict__ vT,
             int N, float* __restrict__ O, _Float16* __restrict__ O16,
             int Dtot, float scale) {
  constexpr int P  = DH / 16;      // output col groups / K staging parts
  constexpr int KS = DH + 16;      // Ks row stride (halves), 16B multiple
  constexpr int VS = 48;           // Vt/Ps row stride
  __shared__ __align__(16) _Float16 Ks[32 * KS];
  __shared__ __align__(16) _Float16 Vt[DH * VS];
  __shared__ __align__(16) _Float16 Ps[4 * 16 * VS];

  const int tid = threadIdx.x, lane = tid & 31, wave = tid >> 5;
  const int head = blockIdx.y;
  const int qrow0 = blockIdx.x * 64 + wave * 16;
  const int qoff = head * DH, koff = 256 + head * DH;
  const int ldq = 768;

  // Q fragments, resident for the whole pass
  v16h qf[DH / 32];
  {
    const int row = qrow0 + (lane & 15);
    const int akb = (lane < 16) ? 0 : 8;
    const _Float16* qp = qkv + (size_t)row * ldq + qoff;
#pragma unroll
    for (int s = 0; s < DH / 32; ++s) qf[s] = load_afrag(qp + s * 32, akb);
  }

  v8f o[P] = {};
  float mrun[8], lsum[8];
#pragma unroll
  for (int v = 0; v < 8; ++v) { mrun[v] = -1e30f; lsum[v] = 0.f; }

  for (int kb = 0; kb < N; kb += 32) {
    { // cooperative staging: K row-major, V from pre-transposed vT (both 16B)
      const int key = tid / P, part = tid % P;
      if (key < 32) {
        const _Float16* src = qkv + (size_t)(kb + key) * ldq + koff + part * 16;
        *(v8h*)(&Ks[key * KS + part * 16])     = *(const v8h*)(src);
        *(v8h*)(&Ks[key * KS + part * 16 + 8]) = *(const v8h*)(src + 8);
      }
      if (tid < DH) {   // one thread per V column: 32 keys = 4x 16B
        const _Float16* src = vT + (size_t)(head * DH + tid) * N + kb;
#pragma unroll
        for (int j = 0; j < 4; ++j)
          *(v8h*)(&Vt[tid * VS + j * 8]) = *(const v8h*)(src + j * 8);
      }
    }
    __syncthreads();

    // S = Q K^T for 32 keys (two 16-col groups)
    v8f s0 = {}, s1 = {};
    {
      const int bkb = (lane < 16) ? 0 : 8;
#pragma unroll
      for (int s = 0; s < DH / 32; ++s) {
        v16h k0 = load_bfrag(&Ks[(lane & 15) * KS + s * 32], bkb);
        v16h k1 = load_bfrag(&Ks[(16 + (lane & 15)) * KS + s * 32], bkb);
        s0 = wmma16(qf[s], k0, s0);
        s1 = wmma16(qf[s], k1, s1);
      }
    }

    // online softmax (rows live as VGPR index v, +8 for lanes>=16; the 16
    // lanes of each half hold the 16 columns -> shfl_xor over masks 8..1)
    float alpha[8];
#pragma unroll
    for (int v = 0; v < 8; ++v) {
      float m = fmaxf(s0[v] * scale, s1[v] * scale);
#pragma unroll
      for (int off = 8; off >= 1; off >>= 1) m = fmaxf(m, __shfl_xor(m, off, 32));
      const float mnew = fmaxf(mrun[v], m);
      alpha[v] = __expf(mrun[v] - mnew);
      const float p0 = __expf(s0[v] * scale - mnew);
      const float p1 = __expf(s1[v] * scale - mnew);
      float ssum = p0 + p1;
#pragma unroll
      for (int off = 8; off >= 1; off >>= 1) ssum += __shfl_xor(ssum, off, 32);
      lsum[v] = lsum[v] * alpha[v] + ssum;
      mrun[v] = mnew;
      s0[v] = p0; s1[v] = p1;   // stash P for LDS round-trip
    }
#pragma unroll
    for (int ci = 0; ci < P; ++ci)
#pragma unroll
      for (int v = 0; v < 8; ++v) o[ci][v] *= alpha[v];

    { // C-layout -> A-layout crossover for P through per-wave LDS slab
      _Float16* pw = &Ps[wave * 16 * VS];
      const int r0 = (lane >= 16) ? 8 : 0, col = lane & 15;
#pragma unroll
      for (int v = 0; v < 8; ++v) {
        pw[(r0 + v) * VS + col]      = (_Float16)s0[v];
        pw[(r0 + v) * VS + col + 16] = (_Float16)s1[v];
      }
    }
    __syncthreads();

    { // O += P V
      const int akb = (lane < 16) ? 0 : 8;
      const int bkb = (lane < 16) ? 0 : 8;
      v16h pf = load_afrag(&Ps[wave * 16 * VS + (lane & 15) * VS], akb);
#pragma unroll
      for (int ci = 0; ci < P; ++ci) {
        v16h bv = load_bfrag(&Vt[(ci * 16 + (lane & 15)) * VS], bkb);
        o[ci] = wmma16(pf, bv, o[ci]);
      }
    }
    __syncthreads();
  }

  const int r0 = (lane >= 16) ? 8 : 0, cl = lane & 15;
#pragma unroll
  for (int ci = 0; ci < P; ++ci)
#pragma unroll
    for (int v = 0; v < 8; ++v) {
      const int row = qrow0 + r0 + v;
      const int col = head * DH + ci * 16 + cl;
      const float val = o[ci][v] / lsum[v];
      O[(size_t)row * Dtot + col] = val;
      if (O16) O16[(size_t)row * Dtot + col] = (_Float16)val;
    }
}

// ---------------------------------------------------------------------------
// Small support kernels
// ---------------------------------------------------------------------------
__global__ void cvt_k(const float* __restrict__ s, _Float16* __restrict__ d, int n) {
  int i = blockIdx.x * 256 + threadIdx.x;
  if (i < n) d[i] = (_Float16)s[i];
}
// f32 [K][Nc] weight -> f16 TRANSPOSED [Nc][K]
__global__ void cvt_t_k(const float* __restrict__ s, _Float16* __restrict__ d,
                        int K, int Nc) {
  int i = blockIdx.x * 256 + threadIdx.x;
  if (i < K * Nc) {
    int k = i / Nc, n = i - k * Nc;
    d[(size_t)n * K + k] = (_Float16)s[i];
  }
}
// transpose the V section of qkv16 [N][768] -> vT [256][N]
__global__ void __launch_bounds__(256)
vtrans_k(const _Float16* __restrict__ qkv, _Float16* __restrict__ vT, int N) {
  const int c = blockIdx.x;             // 0..255 (v column)
  for (int key = threadIdx.x; key < N; key += 256)
    vT[(size_t)c * N + key] = qkv[(size_t)key * 768 + 512 + c];
}
__global__ void zero_k(float* __restrict__ p, int n) {
  int i = blockIdx.x * 256 + threadIdx.x;
  if (i < n) p[i] = 0.f;
}
__global__ void scale_cvt_k(const float* __restrict__ s, float sc,
                            float* __restrict__ d32, _Float16* __restrict__ d16, int n) {
  int i = blockIdx.x * 256 + threadIdx.x;
  if (i < n) { float v = s[i] * sc; if (d32) d32[i] = v; d16[i] = (_Float16)v; }
}

__global__ void __launch_bounds__(256)
layernorm_k(const float* __restrict__ X, const float* __restrict__ g,
            const float* __restrict__ b, float* __restrict__ Y,
            _Float16* __restrict__ Y16, float* __restrict__ accum) {
  __shared__ float red[256];
  const int row = blockIdx.x, tid = threadIdx.x;
  const float x = X[(size_t)row * kD + tid];
  red[tid] = x; __syncthreads();
  for (int s = 128; s > 0; s >>= 1) { if (tid < s) red[tid] += red[tid + s]; __syncthreads(); }
  const float mu = red[0] / kD; __syncthreads();
  const float d = x - mu;
  red[tid] = d * d; __syncthreads();
  for (int s = 128; s > 0; s >>= 1) { if (tid < s) red[tid] += red[tid + s]; __syncthreads(); }
  const float var = red[0] / kD;
  const float y = d * rsqrtf(var + 1e-5f) * g[tid] + b[tid];
  Y[(size_t)row * kD + tid] = y;
  Y16[(size_t)row * kD + tid] = (_Float16)y;
  if (accum) accum[(size_t)row * kD + tid] += y;
}

// edge attention logits: leaky_relu(concat(h[dst],h[src]) . e_w + e_b, 0.2)
__global__ void __launch_bounds__(256)
edge_logits_k(const _Float16* __restrict__ h16, const int* __restrict__ dst,
              const int* __restrict__ src, const float* __restrict__ ew,
              const float* __restrict__ eb, float* __restrict__ logits,
              float* __restrict__ pmax) {
  __shared__ float red[256];
  const int e = blockIdx.x * 256 + threadIdx.x;
  const _Float16* hd = h16 + (size_t)dst[e] * kD;
  const _Float16* hs = h16 + (size_t)src[e] * kD;
  float acc = eb[0];
#pragma unroll 4
  for (int k = 0; k < kD; ++k) acc += (float)hd[k] * ew[k];
#pragma unroll 4
  for (int k = 0; k < kD; ++k) acc += (float)hs[k] * ew[kD + k];
  const float l = acc > 0.f ? acc : 0.2f * acc;
  logits[e] = l;
  red[threadIdx.x] = l; __syncthreads();
  for (int s = 128; s > 0; s >>= 1) {
    if (threadIdx.x < s) red[threadIdx.x] = fmaxf(red[threadIdx.x], red[threadIdx.x + s]);
    __syncthreads();
  }
  if (threadIdx.x == 0) pmax[blockIdx.x] = red[0];
}
__global__ void __launch_bounds__(512)
reduce_max_k(const float* __restrict__ pm, float* __restrict__ out) {
  __shared__ float s[512];
  const int t = threadIdx.x;
  s[t] = pm[t]; __syncthreads();
  for (int k = 256; k > 0; k >>= 1) { if (t < k) s[t] = fmaxf(s[t], s[t + k]); __syncthreads(); }
  if (t == 0) { out[0] = s[0]; out[1] = 0.f; }
}
__global__ void __launch_bounds__(256)
exp_sum_k(float* __restrict__ logits, const float* __restrict__ mx, float* __restrict__ gsum) {
  __shared__ float red[256];
  const int e = blockIdx.x * 256 + threadIdx.x;
  const float v = __expf(logits[e] - mx[0]);
  logits[e] = v;
  red[threadIdx.x] = v; __syncthreads();
  for (int s = 128; s > 0; s >>= 1) { if (threadIdx.x < s) red[threadIdx.x] += red[threadIdx.x + s]; __syncthreads(); }
  if (threadIdx.x == 0) atomicAdd(gsum, red[0]);
}
__global__ void norm_att_k(float* __restrict__ att, const float* __restrict__ gsum) {
  const int e = blockIdx.x * 256 + threadIdx.x;
  att[e] = att[e] / gsum[0];
}

// global mean pool (batch = contiguous blocks of N/B rows per graph)
__global__ void __launch_bounds__(256)
pool_k(const float* __restrict__ X, float* __restrict__ pooled, int rowsPer) {
  const int g = blockIdx.x, c = threadIdx.x;
  float s = 0.f;
  for (int r = 0; r < rowsPer; ++r) s += X[(size_t)(g * rowsPer + r) * kD + c];
  pooled[g * kD + c] = s / (float)rowsPer;
}
__global__ void __launch_bounds__(128)
classifier_k(const float* __restrict__ pooled, const float* __restrict__ w1,
             const float* __restrict__ b1, const float* __restrict__ w2,
             const float* __restrict__ b2, float* __restrict__ out) {
  __shared__ float sh[kB][kD / 2];
  const int c = threadIdx.x;
  for (int g = 0; g < kB; ++g) {
    float a = b1[c];
    for (int k = 0; k < kD; ++k) a += pooled[g * kD + k] * w1[k * (kD / 2) + c];
    sh[g][c] = a > 0.f ? a : 0.f;
  }
  __syncthreads();
  if (c < kB) {
    float a = b2[0];
    for (int k = 0; k < kD / 2; ++k) a += sh[c][k] * w2[k];
    out[c] = a;
  }
}

// ---------------------------------------------------------------------------
// Host orchestration
// ---------------------------------------------------------------------------
static inline char* wsa(char*& p, size_t bytes) {
  char* r = p;
  p += (bytes + 255) & ~(size_t)255;
  return r;
}

static void gemm(hipStream_t st, int act, bool scatter,
                 const _Float16* X, const _Float16* Y, int Ksplit,
                 const int* i0, const int* i1, const _Float16* Bt,
                 const float* bias, const float* rowscale, const float* residual,
                 float* C, _Float16* C16, const int* sidx, int M, int K, int Nc) {
  dim3 g(Nc / 64, M / 64), b(128);
  if (scatter) {
    if (act == 1) gemm_wmma_k<1, true><<<g, b, 0, st>>>(X, Y, Ksplit, i0, i1, Bt, bias, rowscale, residual, C, C16, sidx, M, K, Nc);
    else          gemm_wmma_k<0, true><<<g, b, 0, st>>>(X, Y, Ksplit, i0, i1, Bt, bias, rowscale, residual, C, C16, sidx, M, K, Nc);
  } else {
    if (act == 1) gemm_wmma_k<1, false><<<g, b, 0, st>>>(X, Y, Ksplit, i0, i1, Bt, bias, rowscale, residual, C, C16, sidx, M, K, Nc);
    else          gemm_wmma_k<0, false><<<g, b, 0, st>>>(X, Y, Ksplit, i0, i1, Bt, bias, rowscale, residual, C, C16, sidx, M, K, Nc);
  }
}
static void cvt(hipStream_t st, const float* s, _Float16* d, int n) {
  cvt_k<<<(n + 255) / 256, 256, 0, st>>>(s, d, n);
}
static void cvtT(hipStream_t st, const float* s, _Float16* d, int K, int Nc) {
  cvt_t_k<<<(K * Nc + 255) / 256, 256, 0, st>>>(s, d, K, Nc);
}

extern "C" void kernel_launch(void* const* d_in, const int* in_sizes, int n_in,
                              void* d_out, int out_size, void* d_ws, size_t ws_size,
                              hipStream_t stream) {
  (void)in_sizes; (void)n_in; (void)out_size; (void)ws_size;
  // ---- inputs (setup_inputs order) ----
  const float* x      = (const float*)d_in[0];
  const int*   eidx   = (const int*)d_in[1];
  const int*   esrc   = eidx;          // edge_index[0] = src (x_j)
  const int*   edst   = eidx + kE;     // edge_index[1] = dst (x_i)
  const float* w_in   = (const float*)d_in[3];
  const float* b_in   = (const float*)d_in[4];
  const float* a_inw  = (const float*)d_in[5];
  const float* a_inb  = (const float*)d_in[6];
  const float* a_outw = (const float*)d_in[7];
  const float* a_outb = (const float*)d_in[8];
  const float* m_w    = (const float*)d_in[9];
  const float* m_b    = (const float*)d_in[10];
  const float* u_w    = (const float*)d_in[11];
  const float* u_b    = (const float*)d_in[12];
  const float* e_w    = (const float*)d_in[13];
  const float* e_b    = (const float*)d_in[14];
  const float* ln_g   = (const float*)d_in[15];
  const float* ln_b   = (const float*)d_in[16];
  const float* la_inw = (const float*)d_in[17];
  const float* la_inb = (const float*)d_in[18];
  const float* la_outw= (const float*)d_in[19];
  const float* la_outb= (const float*)d_in[20];
  const float* ga_inw = (const float*)d_in[21];
  const float* ga_inb = (const float*)d_in[22];
  const float* ga_outw= (const float*)d_in[23];
  const float* ga_outb= (const float*)d_in[24];
  const float* c_w    = (const float*)d_in[25];
  const float* c_b    = (const float*)d_in[26];
  const float* c1_w   = (const float*)d_in[27];
  const float* c1_b   = (const float*)d_in[28];
  const float* c2_w   = (const float*)d_in[29];
  const float* c2_b   = (const float*)d_in[30];
  float* out = (float*)d_out;

  // ---- workspace layout ----
  char* p = (char*)d_ws;
  _Float16* x16    = (_Float16*)wsa(p, (size_t)kN * kF * 2);
  float*    h32    = (float*)   wsa(p, (size_t)kN * kD * 4);
  _Float16* h16    = (_Float16*)wsa(p, (size_t)kN * kD * 2);
  float*    qkv32  = (float*)   wsa(p, (size_t)kN * 3 * kD * 4);
  _Float16* qkv16  = (_Float16*)wsa(p, (size_t)kN * 3 * kD * 2);
  _Float16* vT16   = (_Float16*)wsa(p, (size_t)kN * kD * 2);
  float*    o32    = (float*)   wsa(p, (size_t)kN * kD * 4);
  _Float16* o16    = (_Float16*)wsa(p, (size_t)kN * kD * 2);
  float*    agg32  = (float*)   wsa(p, (size_t)kN * kD * 4);
  _Float16* agg16  = (_Float16*)wsa(p, (size_t)kN * kD * 2);
  float*    tmp32  = (float*)   wsa(p, (size_t)kN * kD * 4);
  float*    lsum   = (float*)   wsa(p, (size_t)kN * kD * 4);
  float*    g32    = (float*)   wsa(p, (size_t)kN * kD * 4);
  _Float16* g16    = (_Float16*)wsa(p, (size_t)kN * kD * 2);
  float*    loc32  = (float*)   wsa(p, (size_t)kN * kD * 4);
  _Float16* loc16  = (_Float16*)wsa(p, (size_t)kN * kD * 2);
  float*    glb32  = (float*)   wsa(p, (size_t)kN * kD * 4);
  _Float16* glb16  = (_Float16*)wsa(p, (size_t)kN * kD * 2);
  float*    hc32   = (float*)   wsa(p, (size_t)kN * kD * 4);
  float*    logits = (float*)   wsa(p, (size_t)kE * 4);
  float*    red    = (float*)   wsa(p, (size_t)516 * 4);   // [0..511]=pmax, 512=gmax, 513=gsum
  float*    pooled = (float*)   wsa(p, (size_t)kB * kD * 4);
  // pre-transposed f16 weights ([Nc][K] each)
  _Float16* w_in16   = (_Float16*)wsa(p, (size_t)kF * kD * 2);
  _Float16* a_inw16  = (_Float16*)wsa(p, (size_t)kL * kD * 3 * kD * 2);
  _Float16* a_outw16 = (_Float16*)wsa(p, (size_t)kL * kD * kD * 2);
  _Float16* m_w16    = (_Float16*)wsa(p, (size_t)kL * 2 * kD * kD * 2);
  _Float16* u_w16    = (_Float16*)wsa(p, (size_t)kL * 2 * kD * kD * 2);
  _Float16* la_inw16 = (_Float16*)wsa(p, (size_t)kD * 3 * kD * 2);
  _Float16* la_outw16= (_Float16*)wsa(p, (size_t)kD * kD * 2);
  _Float16* ga_inw16 = (_Float16*)wsa(p, (size_t)kD * 3 * kD * 2);
  _Float16* ga_outw16= (_Float16*)wsa(p, (size_t)kD * kD * 2);
  _Float16* c_w16    = (_Float16*)wsa(p, (size_t)2 * kD * kD * 2);

  const float inv_sqrt32 = 0.17677669529663687f;  // 1/sqrt(32)  H=8
  const float inv_sqrt64 = 0.125f;                // 1/sqrt(64)  H=4

  // ---- conversions: activations plain, weights transposed ----
  cvt(stream, x, x16, kN * kF);
  cvtT(stream, w_in, w_in16, kF, kD);
  for (int l = 0; l < kL; ++l) {
    cvtT(stream, a_inw  + (size_t)l * kD * 3 * kD,  a_inw16  + (size_t)l * kD * 3 * kD,  kD,     3 * kD);
    cvtT(stream, a_outw + (size_t)l * kD * kD,      a_outw16 + (size_t)l * kD * kD,      kD,     kD);
    cvtT(stream, m_w    + (size_t)l * 2 * kD * kD,  m_w16    + (size_t)l * 2 * kD * kD,  2 * kD, kD);
    cvtT(stream, u_w    + (size_t)l * 2 * kD * kD,  u_w16    + (size_t)l * 2 * kD * kD,  2 * kD, kD);
  }
  cvtT(stream, la_inw,  la_inw16,  kD, 3 * kD);
  cvtT(stream, la_outw, la_outw16, kD, kD);
  cvtT(stream, ga_inw,  ga_inw16,  kD, 3 * kD);
  cvtT(stream, ga_outw, ga_outw16, kD, kD);
  cvtT(stream, c_w,     c_w16,     2 * kD, kD);

  // ---- input projection: h = relu(x @ w_in + b_in) ----
  gemm(stream, 1, false, x16, nullptr, kF, nullptr, nullptr, w_in16, b_in,
       nullptr, nullptr, h32, h16, nullptr, kN, kF, kD);
  zero_k<<<(kN * kD + 255) / 256, 256, 0, stream>>>(lsum, kN * kD);

  // ---- 4 graph-transformer layers ----
  for (int l = 0; l < kL; ++l) {
    // global MHA (h replaced, no residual inside mha)
    gemm(stream, 0, false, h16, nullptr, kD, nullptr, nullptr,
         a_inw16 + (size_t)l * kD * 3 * kD, a_inb + (size_t)l * 3 * kD,
         nullptr, nullptr, qkv32, qkv16, nullptr, kN, kD, 3 * kD);
    vtrans_k<<<kD, 256, 0, stream>>>(qkv16, vT16, kN);
    flash_attn_k<32><<<dim3(kN / 64, kH), 128, 0, stream>>>(qkv16, vT16, kN, o32, o16, kD, inv_sqrt32);
    gemm(stream, 0, false, o16, nullptr, kD, nullptr, nullptr,
         a_outw16 + (size_t)l * kD * kD, a_outb + (size_t)l * kD,
         nullptr, nullptr, h32, h16, nullptr, kN, kD, kD);

    // edge softmax attention over ALL edges
    edge_logits_k<<<kE / 256, 256, 0, stream>>>(h16, edst, esrc,
        e_w + (size_t)l * 2 * kD, e_b + l, logits, red);
    reduce_max_k<<<1, 512, 0, stream>>>(red, red + 512);
    exp_sum_k<<<kE / 256, 256, 0, stream>>>(logits, red + 512, red + 513);
    norm_att_k<<<kE / 256, 256, 0, stream>>>(logits, red + 513);

    // msg = relu(concat(h[dst],h[src]) @ m_w + m_b) * att ; segment_sum -> agg
    zero_k<<<(kN * kD + 255) / 256, 256, 0, stream>>>(agg32, kN * kD);
    gemm(stream, 1, true, h16, h16, kD, edst, esrc,
         m_w16 + (size_t)l * 2 * kD * kD, m_b + (size_t)l * kD,
         logits, nullptr, agg32, nullptr, edst, kE, 2 * kD, kD);
    cvt(stream, agg32, agg16, kN * kD);

    // out = relu(concat(h,agg) @ u_w + u_b) + h ; layernorm (+= into lsum)
    gemm(stream, 1, false, h16, agg16, kD, nullptr, nullptr,
         u_w16 + (size_t)l * 2 * kD * kD, u_b + (size_t)l * kD,
         nullptr, h32, tmp32, nullptr, nullptr, kN, 2 * kD, kD);
    layernorm_k<<<kN, 256, 0, stream>>>(tmp32, ln_g + (size_t)l * kD,
                                        ln_b + (size_t)l * kD, h32, h16, lsum);
  }

  // ---- multi-scale attention: local (H=8) on h ----
  gemm(stream, 0, false, h16, nullptr, kD, nullptr, nullptr, la_inw16, la_inb,
       nullptr, nullptr, qkv32, qkv16, nullptr, kN, kD, 3 * kD);
  vtrans_k<<<kD, 256, 0, stream>>>(qkv16, vT16, kN);
  flash_attn_k<32><<<dim3(kN / 64, kH), 128, 0, stream>>>(qkv16, vT16, kN, o32, o16, kD, inv_sqrt32);
  gemm(stream, 0, false, o16, nullptr, kD, nullptr, nullptr, la_outw16, la_outb,
       nullptr, nullptr, loc32, loc16, nullptr, kN, kD, kD);

  // ---- global branch (H=4, dh=64) on mean of layer outputs ----
  scale_cvt_k<<<(kN * kD + 255) / 256, 256, 0, stream>>>(lsum, 1.f / kL, g32, g16, kN * kD);
  gemm(stream, 0, false, g16, nullptr, kD, nullptr, nullptr, ga_inw16, ga_inb,
       nullptr, nullptr, qkv32, qkv16, nullptr, kN, kD, 3 * kD);
  vtrans_k<<<kD, 256, 0, stream>>>(qkv16, vT16, kN);
  flash_attn_k<64><<<dim3(kN / 64, kH / 2), 128, 0, stream>>>(qkv16, vT16, kN, o32, o16, kD, inv_sqrt64);
  gemm(stream, 0, false, o16, nullptr, kD, nullptr, nullptr, ga_outw16, ga_outb,
       nullptr, nullptr, glb32, glb16, nullptr, kN, kD, kD);

  // ---- combine: relu(concat(local, glob) @ c_w + c_b) ----
  gemm(stream, 1, false, loc16, glb16, kD, nullptr, nullptr, c_w16, c_b,
       nullptr, nullptr, hc32, nullptr, nullptr, kN, 2 * kD, kD);

  // ---- pool + classifier ----
  pool_k<<<kB, 256, 0, stream>>>(hc32, pooled, kN / kB);
  classifier_k<<<1, 128, 0, stream>>>(pooled, c1_w, c1_b, c2_w, c2_b, out);
}
